// SlotAttention_31336081391885
// MI455X (gfx1250) — matrix operand order repaired and checked by reading, not compiled, not run on hardware
//
#include <hip/hip_runtime.h>

// ---------------------------------------------------------------------------
// Slot Attention forward for MI455X (gfx1250, wave32, WMMA + TDM).
// f16 storage + f32 WMMA accumulation for all GEMMs; slots padded S=8 -> M=16
// with hard-zero pad rows; softmax reads only real rows. Input tiles staged
// into LDS with the Tensor Data Mover; updates GEMM uses double-buffered LDS
// transpose staging (one barrier per K-step).
// ---------------------------------------------------------------------------

#define DEV static __device__ __forceinline__

typedef _Float16 h16;
typedef __attribute__((ext_vector_type(16))) _Float16 v16h;
typedef __attribute__((ext_vector_type(8)))  _Float16 v8h;
typedef __attribute__((ext_vector_type(8)))  float    v8f;
typedef __attribute__((ext_vector_type(4)))  unsigned int u32x4;
typedef __attribute__((ext_vector_type(8)))  unsigned int u32x8;

#define BATCH 64
#define NPOS  4096
#define DIM   256
#define NSLOT 8
#define HID   512

// ---- WMMA helpers (layouts per CDNA5 ISA) ----------------------

DEV v8f wmma_f16(v16h a, v16h b, v8f c) {
  return __builtin_amdgcn_wmma_f32_16x16x32_f16(
      /*neg_a=*/false, a, /*neg_b=*/false, b,
      /*c_mod=*/(short)0, c, /*reuse_a=*/false, /*reuse_b=*/false);
}

// A fragment: 16x32 f16 tile, row-major, row stride ld (halves).
// lane<16: K = {0..7, 16..23}; lane>=16: K = {8..15, 24..31}.
DEV v16h load_a_frag(const h16* tile, int ld, int lane) {
  const int m = lane & 15, hf = lane >> 4;
  const h16* p = tile + m * ld + hf * 8;
  v8h lo = *(const v8h*)(p);
  v8h hi = *(const v8h*)(p + 16);
  return __builtin_shufflevector(lo, hi, 0,1,2,3,4,5,6,7,8,9,10,11,12,13,14,15);
}

// B fragment for C = A @ W^T with W row-major [out][in]:
// B[k][n] = W[n][k]; lane column n = lane&15 holds K = {hf*8..hf*8+7, 16+hf*8..16+hf*8+7}.
DEV v16h load_bT_frag(const h16* w, int ld, int lane) {
  const int n = lane & 15, hf = lane >> 4;
  const h16* p = w + n * ld + hf * 8;
  v8h lo = *(const v8h*)(p);
  v8h hi = *(const v8h*)(p + 16);
  return __builtin_shufflevector(lo, hi, 0,1,2,3,4,5,6,7,8,9,10,11,12,13,14,15);
}

// ---- LayerNorm of one 256-wide row by one wave32; writes f16 --------------
DEV void ln_row_f16(const float* x, h16* dst, const float* gw, const float* gb,
                    int lane) {
  float v[8];
  float s = 0.f, s2 = 0.f;
#pragma unroll
  for (int i = 0; i < 8; ++i) {
    v[i] = x[lane * 8 + i];
    s += v[i];
    s2 += v[i] * v[i];
  }
#pragma unroll
  for (int off = 16; off > 0; off >>= 1) {
    s  += __shfl_xor(s,  off, 32);
    s2 += __shfl_xor(s2, off, 32);
  }
  const float mu = s * (1.f / 256.f);
  const float rs = rsqrtf(s2 * (1.f / 256.f) - mu * mu + 1e-5f);
#pragma unroll
  for (int i = 0; i < 8; ++i)
    dst[lane * 8 + i] =
        (h16)((v[i] - mu) * rs * gw[lane * 8 + i] + gb[lane * 8 + i]);
}

// ---------------------------------------------------------------------------
// f32 -> f16 weight conversion
__global__ void sa_conv_f16(const float* __restrict__ s, h16* __restrict__ d,
                            int n) {
  int i = blockIdx.x * 256 + threadIdx.x;
  if (i < n) d[i] = (h16)s[i];
}

// slots[b][i][d] = query_pos[i][d] for i<8 else 0  (padded M=16 layout)
__global__ void sa_init_slots(const float* __restrict__ qp,
                              float* __restrict__ slots) {
  int i = blockIdx.x * 256 + threadIdx.x;   // over 64*16*256
  int d = i & 255, row = (i >> 8) & 15;
  slots[i] = (row < NSLOT) ? qp[row * DIM + d] : 0.f;
}

// ---------------------------------------------------------------------------
// Fused LN(inputs) + k/v projection. 32 input rows per block, 8 waves.
// Input tile (32x256 f32, 32KB) is DMA'd into LDS by the Tensor Data Mover
// (one 2-D D#: tile 256x32 elements of 4B, stride 256), then LN reads LDS.
__global__ __launch_bounds__(256) void sa_kv(
    const float* __restrict__ inputs, const float* __restrict__ lnw,
    const float* __restrict__ lnb, const h16* __restrict__ wkh,
    const float* __restrict__ bk, const h16* __restrict__ wvh,
    const float* __restrict__ bv, h16* __restrict__ kh, h16* __restrict__ vh) {
  __shared__ __align__(16) float xs[32 * 256];  // TDM destination (f32 tile)
  __shared__ __align__(32) h16 xh[32 * 256];    // LN'd f16 tile
  const int tid = threadIdx.x, wave = tid >> 5, lane = tid & 31;
  const long row0 = (long)blockIdx.x * 32;

  if (wave == 0) {
    // Tensor DMA descriptor (CDNA5 ISA).
    const unsigned long long ga =
        (unsigned long long)(const void*)(inputs + row0 * DIM);
    const unsigned int lds = (unsigned int)(unsigned long long)(void*)&xs[0];
    u32x4 g0;
    g0[0] = 1u;                                   // count=1 user descriptor
    g0[1] = lds;                                  // lds_addr (bytes)
    g0[2] = (unsigned int)ga;                     // global_addr[31:0]
    g0[3] = ((unsigned int)(ga >> 32) & 0x01ffffffu) | (2u << 30);  // type=2
    u32x8 g1;
    g1[0] = (2u << 16);                           // data_size = 4B
    g1[1] = (256u << 16);                         // tensor_dim0 = 256 (lo16)
    g1[2] = (32u << 16);                          // td0 hi=0 | tensor_dim1=32
    g1[3] = (256u << 16);                         // td1 hi=0 | tile_dim0=256
    g1[4] = 32u;                                  // tile_dim1=32, tile_dim2=0
    g1[5] = 256u;                                 // tensor_dim0_stride lo32
    g1[6] = 0u;
    g1[7] = 0u;
    asm volatile("tensor_load_to_lds %0, %1" :: "s"(g0), "s"(g1) : "memory");
    __builtin_amdgcn_s_wait_tensorcnt(0);
  }
  __syncthreads();

  for (int rr = 0; rr < 4; ++rr) {
    int r = wave + rr * 8;
    ln_row_f16(xs + r * 256, xh + r * 256, lnw, lnb, lane);
  }
  __syncthreads();

  const int n = lane & 15, hf = lane >> 4;
  for (int t = wave; t < 64; t += 8) {
    const int mat = t >> 5, wi = t & 31, mt = wi >> 4, nt = wi & 15;
    const h16* W = (mat ? wvh : wkh) + nt * 16 * 256;
    const float* bias = (mat ? bv : bk) + nt * 16;
    h16* out = mat ? vh : kh;
    v8f c = {};
#pragma unroll
    for (int k = 0; k < 8; ++k) {
      v16h a = load_a_frag(xh + mt * 16 * 256 + k * 32, 256, lane);
      v16h b = load_bT_frag(W + k * 32, 256, lane);
      c = wmma_f16(a, b, c);
    }
    const float bj = bias[n];
    const long orow = row0 + mt * 16 + hf * 8;
#pragma unroll
    for (int r = 0; r < 8; ++r)
      out[(orow + r) * DIM + nt * 16 + n] = (h16)(c[r] + bj);
  }
}

// ---------------------------------------------------------------------------
// LN(slots) + q = sln @ wq^T + bq.  One block per batch.
__global__ __launch_bounds__(256) void sa_qproj(
    const float* __restrict__ slots, const float* __restrict__ lnsw,
    const float* __restrict__ lnsb, const h16* __restrict__ wqh,
    const float* __restrict__ bq, h16* __restrict__ qh) {
  __shared__ __align__(32) h16 sln[16 * 256];
  const int tid = threadIdx.x, wave = tid >> 5, lane = tid & 31;
  const int b = blockIdx.x;

  ln_row_f16(slots + ((long)b * 16 + wave) * DIM, sln + wave * 256, lnsw, lnsb,
             lane);
#pragma unroll
  for (int i = 0; i < 8; ++i)  // hard-zero pad rows 8..15
    sln[(8 + wave) * 256 + lane * 8 + i] = (h16)0.f;
  __syncthreads();

  const int n = lane & 15, hf = lane >> 4;
  for (int ti = wave; ti < 16; ti += 8) {
    v8f c = {};
#pragma unroll
    for (int k = 0; k < 8; ++k) {
      v16h a = load_a_frag(sln + k * 32, 256, lane);
      v16h bb = load_bT_frag(wqh + ti * 16 * 256 + k * 32, 256, lane);
      c = wmma_f16(a, bb, c);
    }
    const float bj = bq[ti * 16 + n];
#pragma unroll
    for (int r = 0; r < 8; ++r)
      qh[((long)b * 16 + hf * 8 + r) * DIM + ti * 16 + n] = (h16)(c[r] + bj);
  }
}

// ---------------------------------------------------------------------------
// dots[b][i][j] = SCALE * sum_d q[b,i,d] * k[b,j,d].  M=16 (8 real), N=4096,
// K=256. grid (32, B); wave w owns columns j0 = bx*128 + w*16.
__global__ __launch_bounds__(256) void sa_dots(const h16* __restrict__ qh,
                                               const h16* __restrict__ kh,
                                               float* __restrict__ dst) {
  const int tid = threadIdx.x, wave = tid >> 5, lane = tid & 31;
  const int b = blockIdx.y;
  const int j0 = blockIdx.x * 128 + wave * 16;
  const h16* qb = qh + (long)b * 16 * DIM;
  const h16* kb = kh + (long)b * NPOS * DIM;
  v8f c = {};
#pragma unroll
  for (int k = 0; k < 8; ++k) {
    v16h a = load_a_frag(qb + k * 32, 256, lane);
    v16h bb = load_bT_frag(kb + (long)j0 * DIM + k * 32, 256, lane);
    c = wmma_f16(a, bb, c);
  }
  const int n = lane & 15, hf = lane >> 4;
  if (hf == 0) {  // only the 8 real slot rows are kept
#pragma unroll
    for (int r = 0; r < 8; ++r)
      dst[((long)b * NSLOT + r) * NPOS + j0 + n] = c[r] * 0.0625f;
  }
}

// ---------------------------------------------------------------------------
// softmax over slot axis per column j, + EPS, then renormalize over N.
// One block per batch; two passes over dots; writes f16 attn (pad rows = 0).
__global__ __launch_bounds__(256) void sa_softmax(const float* __restrict__ dt,
                                                  h16* __restrict__ attnh) {
  __shared__ float rs[NSLOT];
  const int tid = threadIdx.x, b = blockIdx.x;
  const float* db = dt + (long)b * NSLOT * NPOS;
  h16* ab = attnh + (long)b * 16 * NPOS;
  if (tid < NSLOT) rs[tid] = 0.f;
  __syncthreads();

  float acc[NSLOT] = {0.f, 0.f, 0.f, 0.f, 0.f, 0.f, 0.f, 0.f};
  for (int j = tid; j < NPOS; j += 256) {
    float v[NSLOT], m = -3.4e38f;
#pragma unroll
    for (int i = 0; i < NSLOT; ++i) { v[i] = db[i * NPOS + j]; m = fmaxf(m, v[i]); }
    float s = 0.f;
#pragma unroll
    for (int i = 0; i < NSLOT; ++i) { v[i] = __expf(v[i] - m); s += v[i]; }
    const float inv = 1.f / s;
#pragma unroll
    for (int i = 0; i < NSLOT; ++i) acc[i] += v[i] * inv + 1e-8f;
  }
#pragma unroll
  for (int i = 0; i < NSLOT; ++i) atomicAdd(&rs[i], acc[i]);
  __syncthreads();

  float tot[NSLOT];
#pragma unroll
  for (int i = 0; i < NSLOT; ++i) tot[i] = 1.f / rs[i];

  for (int j = tid; j < NPOS; j += 256) {
    float v[NSLOT], m = -3.4e38f;
#pragma unroll
    for (int i = 0; i < NSLOT; ++i) { v[i] = db[i * NPOS + j]; m = fmaxf(m, v[i]); }
    float s = 0.f;
#pragma unroll
    for (int i = 0; i < NSLOT; ++i) { v[i] = __expf(v[i] - m); s += v[i]; }
    const float inv = 1.f / s;
#pragma unroll
    for (int i = 0; i < NSLOT; ++i)
      ab[i * NPOS + j] = (h16)((v[i] * inv + 1e-8f) * tot[i]);
#pragma unroll
    for (int i = 0; i < NSLOT; ++i) ab[(NSLOT + i) * NPOS + j] = (h16)0.f;
  }
}

// ---------------------------------------------------------------------------
// updates[b][i][d] = sum_j attn[b,i,j] * v[b,j,d].  M=16, N=256, K=4096.
// grid (2, B): block owns 128 d-columns; v tiles transposed through a
// DOUBLE-BUFFERED LDS staging area -> one barrier per K-step.
__global__ __launch_bounds__(256) void sa_updates(const h16* __restrict__ attnh,
                                                  const h16* __restrict__ vh,
                                                  float* __restrict__ upd) {
  __shared__ __align__(32) h16 vT[2][128 * 32];  // [buf][dLocal][j]
  const int tid = threadIdx.x, wave = tid >> 5, lane = tid & 31;
  const int b = blockIdx.y, d0 = blockIdx.x * 128;
  const h16* ab = attnh + (long)b * 16 * NPOS;
  const h16* vb = vh + (long)b * NPOS * DIM;

  const int jl = tid >> 3, ch = (tid & 7) * 16;
  v8f c = {};
  for (int kt = 0; kt < 128; ++kt) {
    const int p = kt & 1;
    const int j0 = kt * 32;
    const h16* src = vb + (long)(j0 + jl) * DIM + d0 + ch;
    if (kt + 1 < 128)
      __builtin_prefetch(vb + (long)(j0 + 32 + jl) * DIM + d0 + ch, 0, 1);
    v8h p0 = *(const v8h*)(src);
    v8h p1 = *(const v8h*)(src + 8);
#pragma unroll
    for (int i = 0; i < 8; ++i) {
      vT[p][(ch + i) * 32 + jl]     = p0[i];
      vT[p][(ch + 8 + i) * 32 + jl] = p1[i];
    }
    __syncthreads();  // single barrier is safe with ping-pong buffers
    v16h a = load_a_frag(ab + j0, NPOS, lane);
    v16h bb = load_bT_frag(vT[p] + (wave * 16) * 32, 32, lane);
    c = wmma_f16(a, bb, c);
  }
  const int n = lane & 15, hf = lane >> 4;
#pragma unroll
  for (int r = 0; r < 8; ++r)  // pad rows come out exactly 0 (attn pad = 0)
    upd[((long)b * 16 + hf * 8 + r) * DIM + d0 + wave * 16 + n] = c[r];
}

// ---------------------------------------------------------------------------
// Fused GRU cell + LN + FFN, one block per batch. Gate r/z GEMMs chain the
// x@W_ih^T and h@W_hh^T products into one WMMA accumulator.
__global__ __launch_bounds__(256) void sa_gruff(
    const float* __restrict__ upd, float* __restrict__ slots,
    const h16* __restrict__ wih, const float* __restrict__ bih,
    const h16* __restrict__ whh, const float* __restrict__ bhh,
    const h16* __restrict__ w1h, const float* __restrict__ b1,
    const h16* __restrict__ w2h, const float* __restrict__ b2,
    const float* __restrict__ lnfw, const float* __restrict__ lnfb) {
  __shared__ __align__(32) char smem[49152];
  h16* uh = (h16*)smem;                    // [16*256] updates f16
  h16* sh = uh + 16 * 256;                 // [16*256] slots f16
  float* gates = (float*)(smem + 16384);   // [8*1024]: rz | xn | hn
  // phase-2 overlays:
  float* hbuf = (float*)smem;              // [8*256]   (over uh)
  h16* lnh = (h16*)(smem + 8192);          // [16*256]  (over sh)
  h16* fbuf = (h16*)(smem + 16384);        // [16*512]  (over gates)

  const int tid = threadIdx.x, wave = tid >> 5, lane = tid & 31;
  const int b = blockIdx.x;
  const float* ub = upd + (long)b * 16 * DIM;
  float* sb = slots + (long)b * 16 * DIM;

  for (int e = tid; e < 16 * 256; e += 256) {
    uh[e] = (h16)ub[e];
    sh[e] = (h16)sb[e];
  }
  __syncthreads();

  const int n = lane & 15, hf = lane >> 4;
  // ---- gate GEMMs: 64 output tiles across 8 waves ----
  for (int t = wave; t < 64; t += 8) {
    v8f c = {};
    if (t < 32) {  // r,z: sum of both projections in one accumulator
      const int j0 = t * 16;
#pragma unroll
      for (int k = 0; k < 8; ++k)
        c = wmma_f16(load_a_frag(uh + k * 32, 256, lane),
                     load_bT_frag(wih + j0 * 256 + k * 32, 256, lane), c);
#pragma unroll
      for (int k = 0; k < 8; ++k)
        c = wmma_f16(load_a_frag(sh + k * 32, 256, lane),
                     load_bT_frag(whh + j0 * 256 + k * 32, 256, lane), c);
      if (hf == 0) {
        const float bj = bih[j0 + n] + bhh[j0 + n];
#pragma unroll
        for (int r = 0; r < 8; ++r) gates[r * 1024 + j0 + n] = c[r] + bj;
      }
    } else if (t < 48) {  // xn
      const int j0 = 512 + (t - 32) * 16;
#pragma unroll
      for (int k = 0; k < 8; ++k)
        c = wmma_f16(load_a_frag(uh + k * 32, 256, lane),
                     load_bT_frag(wih + j0 * 256 + k * 32, 256, lane), c);
      if (hf == 0) {
        const float bj = bih[j0 + n];
#pragma unroll
        for (int r = 0; r < 8; ++r)
          gates[r * 1024 + 512 + (t - 32) * 16 + n] = c[r] + bj;
      }
    } else {  // hn
      const int j0 = 512 + (t - 48) * 16;
#pragma unroll
      for (int k = 0; k < 8; ++k)
        c = wmma_f16(load_a_frag(sh + k * 32, 256, lane),
                     load_bT_frag(whh + j0 * 256 + k * 32, 256, lane), c);
      if (hf == 0) {
        const float bj = bhh[j0 + n];
#pragma unroll
        for (int r = 0; r < 8; ++r)
          gates[r * 1024 + 768 + (t - 48) * 16 + n] = c[r] + bj;
      }
    }
  }
  __syncthreads();

  // ---- elementwise GRU ----
  for (int e = tid; e < 8 * 256; e += 256) {
    const int i = e >> 8, d = e & 255;
    const float r = 1.f / (1.f + __expf(-gates[i * 1024 + d]));
    const float z = 1.f / (1.f + __expf(-gates[i * 1024 + 256 + d]));
    const float nn = tanhf(gates[i * 1024 + 512 + d] + r * gates[i * 1024 + 768 + d]);
    const float sv = sb[i * 256 + d];
    hbuf[i * 256 + d] = (1.f - z) * nn + z * sv;
  }
  __syncthreads();

  // ---- LN(h) rows, zero pad rows ----
  ln_row_f16(hbuf + wave * 256, lnh + wave * 256, lnfw, lnfb, lane);
#pragma unroll
  for (int i = 0; i < 8; ++i) lnh[(8 + wave) * 256 + lane * 8 + i] = (h16)0.f;
  __syncthreads();

  // ---- f = relu(lnh @ w1^T + b1): N=512, K=256 ----
  for (int t = wave; t < 32; t += 8) {
    const int j0 = t * 16;
    v8f c = {};
#pragma unroll
    for (int k = 0; k < 8; ++k)
      c = wmma_f16(load_a_frag(lnh + k * 32, 256, lane),
                   load_bT_frag(w1h + j0 * 256 + k * 32, 256, lane), c);
    const float bj = b1[j0 + n];
#pragma unroll
    for (int r = 0; r < 8; ++r) {
      const float v = c[r] + bj;
      fbuf[(hf * 8 + r) * 512 + j0 + n] = (h16)(v > 0.f ? v : 0.f);
    }
  }
  __syncthreads();

  // ---- slots = h + f @ w2^T + b2: N=256, K=512 ----
  for (int t = wave; t < 16; t += 8) {
    const int j0 = t * 16;
    v8f c = {};
#pragma unroll
    for (int k = 0; k < 16; ++k)
      c = wmma_f16(load_a_frag(fbuf + k * 32, 512, lane),
                   load_bT_frag(w2h + j0 * 512 + k * 32, 512, lane), c);
    if (hf == 0) {
      const float bj = b2[j0 + n];
#pragma unroll
      for (int r = 0; r < 8; ++r)
        sb[r * 256 + j0 + n] = c[r] + bj + hbuf[r * 256 + j0 + n];
    }
  }
}

// ---------------------------------------------------------------------------
__global__ void sa_writeout(const float* __restrict__ slots,
                            float* __restrict__ out) {
  int i = blockIdx.x * 256 + threadIdx.x;  // 64*8*256
  int d = i & 255, row = (i >> 8) & 7, b = i >> 11;
  out[i] = slots[((long)b * 16 + row) * DIM + d];
}

// ---------------------------------------------------------------------------
extern "C" void kernel_launch(void* const* d_in, const int* in_sizes, int n_in,
                              void* d_out, int out_size, void* d_ws,
                              size_t ws_size, hipStream_t stream) {
  const float* inputs  = (const float*)d_in[0];
  const float* qpos    = (const float*)d_in[1];
  const float* wq      = (const float*)d_in[2];
  const float* bq      = (const float*)d_in[3];
  const float* wk      = (const float*)d_in[4];
  const float* bk      = (const float*)d_in[5];
  const float* wv      = (const float*)d_in[6];
  const float* bv      = (const float*)d_in[7];
  const float* w_ih    = (const float*)d_in[8];
  const float* b_ih    = (const float*)d_in[9];
  const float* w_hh    = (const float*)d_in[10];
  const float* b_hh    = (const float*)d_in[11];
  const float* w1      = (const float*)d_in[12];
  const float* b1      = (const float*)d_in[13];
  const float* w2      = (const float*)d_in[14];
  const float* b2      = (const float*)d_in[15];
  const float* ln_in_w = (const float*)d_in[16];
  const float* ln_in_b = (const float*)d_in[17];
  const float* ln_s_w  = (const float*)d_in[18];
  const float* ln_s_b  = (const float*)d_in[19];
  const float* ln_ff_w = (const float*)d_in[20];
  const float* ln_ff_b = (const float*)d_in[21];

  char* w = (char*)d_ws;
  size_t off = 0;
  auto alloc = [&](size_t bytes) {
    size_t r = off;
    off += (bytes + 255) & ~(size_t)255;
    return r;
  };
  h16* kh    = (h16*)(w + alloc((size_t)BATCH * NPOS * DIM * 2));
  h16* vh    = (h16*)(w + alloc((size_t)BATCH * NPOS * DIM * 2));
  h16* wqh   = (h16*)(w + alloc((size_t)DIM * DIM * 2));
  h16* wkh   = (h16*)(w + alloc((size_t)DIM * DIM * 2));
  h16* wvh   = (h16*)(w + alloc((size_t)DIM * DIM * 2));
  h16* wihh  = (h16*)(w + alloc((size_t)3 * DIM * DIM * 2));
  h16* whhh  = (h16*)(w + alloc((size_t)3 * DIM * DIM * 2));
  h16* w1h   = (h16*)(w + alloc((size_t)HID * DIM * 2));
  h16* w2h   = (h16*)(w + alloc((size_t)DIM * HID * 2));
  float* slots = (float*)(w + alloc((size_t)BATCH * 16 * DIM * 4));
  h16* qh    = (h16*)(w + alloc((size_t)BATCH * 16 * DIM * 2));
  float* dts = (float*)(w + alloc((size_t)BATCH * NSLOT * NPOS * 4));
  h16* attnh = (h16*)(w + alloc((size_t)BATCH * 16 * NPOS * 2));
  float* upd = (float*)(w + alloc((size_t)BATCH * 16 * DIM * 4));

  // weight conversions
  sa_conv_f16<<<(DIM * DIM + 255) / 256, 256, 0, stream>>>(wq, wqh, DIM * DIM);
  sa_conv_f16<<<(DIM * DIM + 255) / 256, 256, 0, stream>>>(wk, wkh, DIM * DIM);
  sa_conv_f16<<<(DIM * DIM + 255) / 256, 256, 0, stream>>>(wv, wvh, DIM * DIM);
  sa_conv_f16<<<(3 * DIM * DIM + 255) / 256, 256, 0, stream>>>(w_ih, wihh, 3 * DIM * DIM);
  sa_conv_f16<<<(3 * DIM * DIM + 255) / 256, 256, 0, stream>>>(w_hh, whhh, 3 * DIM * DIM);
  sa_conv_f16<<<(HID * DIM + 255) / 256, 256, 0, stream>>>(w1, w1h, HID * DIM);
  sa_conv_f16<<<(DIM * HID + 255) / 256, 256, 0, stream>>>(w2, w2h, DIM * HID);

  sa_init_slots<<<(BATCH * 16 * DIM) / 256, 256, 0, stream>>>(qpos, slots);

  sa_kv<<<(BATCH * NPOS) / 32, 256, 0, stream>>>(inputs, ln_in_w, ln_in_b, wkh,
                                                 bk, wvh, bv, kh, vh);

  for (int it = 0; it < 4; ++it) {  // 3 iters + stop-gradient step
    sa_qproj<<<BATCH, 256, 0, stream>>>(slots, ln_s_w, ln_s_b, wqh, bq, qh);
    sa_dots<<<dim3(NPOS / 128, BATCH), 256, 0, stream>>>(qh, kh, dts);
    sa_softmax<<<BATCH, 256, 0, stream>>>(dts, attnh);
    sa_updates<<<dim3(DIM / 128, BATCH), 256, 0, stream>>>(attnh, vh, upd);
    sa_gruff<<<BATCH, 256, 0, stream>>>(upd, slots, wihh, b_ih, whhh, b_hh,
                                        w1h, b1, w2h, b2, ln_ff_w, ln_ff_b);
  }

  sa_writeout<<<(BATCH * NSLOT * DIM) / 256, 256, 0, stream>>>(slots,
                                                               (float*)d_out);
}
